// SS2D_Enhanced_42563125903831
// MI455X (gfx1250) — compile-verified, matrix operand order checked by reading, not run on hardware
//
#include <hip/hip_runtime.h>
#include <hip/hip_bf16.h>
#include <math.h>

typedef __attribute__((ext_vector_type(16))) _Float16 v16h;
typedef __attribute__((ext_vector_type(8)))  _Float16 v8h;
typedef __attribute__((ext_vector_type(8)))  float    v8f;

#define BATCH   4
#define LSEQ    1024
#define DMODEL  256
#define DSTATE  128
#define DCONV   4
#define HEADDIM 64
#define NHEADS  4
#define DINNER  256
#define CONVDIM 512      // DINNER + 2*DSTATE
#define DPROJ   772      // 2*DINNER + 2*DSTATE + NHEADS
#define DPROJ_P 832      // padded to 4*16 tile groups (52 tiles)
#define EPSV    1e-5f

// ---------------------------------------------------------------------------
// helpers
// ---------------------------------------------------------------------------
__device__ inline float block_reduce_sum256(float v, float* sm) {
    // 256 threads = 8 waves (wave32)
    #pragma unroll
    for (int o = 16; o > 0; o >>= 1) v += __shfl_xor(v, o, 32);
    int lane = threadIdx.x & 31, w = threadIdx.x >> 5;
    if (lane == 0) sm[w] = v;
    __syncthreads();
    float r = 0.f;
    if (w == 0) {
        r = (lane < 8) ? sm[lane] : 0.f;
        #pragma unroll
        for (int o = 4; o > 0; o >>= 1) r += __shfl_xor(r, o, 32);
        if (lane == 0) sm[0] = r;
    }
    __syncthreads();
    r = sm[0];
    __syncthreads();
    return r;
}

__device__ inline float silu_f(float x)  { return x / (1.f + __expf(-x)); }
__device__ inline float sigm_f(float x)  { return 1.f / (1.f + __expf(-x)); }
__device__ inline float softplus_f(float x) { return (x > 20.f) ? x : log1pf(__expf(x)); }

// ---------------------------------------------------------------------------
// 0) f32 -> f16 convert with zero padding (weights)
// ---------------------------------------------------------------------------
__global__ void k_cvt_pad_f16(const float* __restrict__ src, _Float16* __restrict__ dst,
                              int n_src, int n_dst) {
    int i = blockIdx.x * blockDim.x + threadIdx.x;
    if (i < n_dst) dst[i] = (i < n_src) ? (_Float16)src[i] : (_Float16)0.f;
}

// ---------------------------------------------------------------------------
// 1) LayerNorm over channels: x(b,c,h,w) -> u_f16(b,l,c)
// ---------------------------------------------------------------------------
__global__ void k_layernorm(const float* __restrict__ x,
                            const float* __restrict__ g,
                            const float* __restrict__ bta,
                            _Float16* __restrict__ u16) {
    __shared__ float sm[8];
    int row = blockIdx.x;                 // b*LSEQ + l
    int b = row >> 10, l = row & 1023;
    int c = threadIdx.x;
    float v = x[((size_t)(b * DMODEL + c) << 10) + l];
    float mu = block_reduce_sum256(v, sm) * (1.f / DMODEL);
    float d  = v - mu;
    float var = block_reduce_sum256(d * d, sm) * (1.f / DMODEL);
    float un = d * rsqrtf(var + EPSV) * g[c] + bta[c];
    u16[(size_t)row * DMODEL + c] = (_Float16)un;
}

// ---------------------------------------------------------------------------
// 2) WMMA f16 GEMM, register-blocked 16x64 per wave (A fragment reused 4x):
//    C(MxNpad, f32) = A(MxK, f16 rowmajor) * B(NpadxK, f16 rowmajor)^T
//    Per ISA 7.12.2 16-bit A/B layout: lane holds matrix row (lane&15);
//    halves 0..7 hold K {0..7 | 8..15} and halves 8..15 hold K {16..23 | 24..31}
//    selected by lane<16 / lane>=16.
// ---------------------------------------------------------------------------
__device__ inline v16h load_frag16(const _Float16* __restrict__ tile, int ld, int lane) {
    const _Float16* p = tile + (size_t)(lane & 15) * ld + ((lane & 16) ? 8 : 0);
    v8h lo = *(const v8h*)(p);
    v8h hi = *(const v8h*)(p + 16);
    v16h f;
    #pragma unroll
    for (int i = 0; i < 8; ++i) { f[i] = lo[i]; f[i + 8] = hi[i]; }
    return f;
}

__global__ void k_wmma_gemm4(const _Float16* __restrict__ A,
                             const _Float16* __restrict__ B,
                             float* __restrict__ C,
                             int K, int ntiles) {
    int lane = threadIdx.x & 31;
    int wave = threadIdx.x >> 5;
    int m_tile = blockIdx.x;
    int ngrp = blockIdx.y * 8 + wave;      // group of 4 consecutive n-tiles
    int n0 = ngrp * 4;
    if (n0 >= ntiles) return;              // whole-wave branch; EXEC all-ones inside

    const _Float16* Abase = A + (size_t)(m_tile * 16) * K;
    const _Float16* B0 = B + (size_t)((n0 + 0) * 16) * K;
    const _Float16* B1 = B + (size_t)((n0 + 1) * 16) * K;
    const _Float16* B2 = B + (size_t)((n0 + 2) * 16) * K;
    const _Float16* B3 = B + (size_t)((n0 + 3) * 16) * K;

    v8f acc0 = {}, acc1 = {}, acc2 = {}, acc3 = {};
    for (int k0 = 0; k0 < K; k0 += 32) {
        __builtin_prefetch(B0 + k0 + 64, 0, 0);   // global_prefetch_b8
        __builtin_prefetch(B2 + k0 + 64, 0, 0);
        v16h af  = load_frag16(Abase + k0, K, lane);
        v16h bf0 = load_frag16(B0 + k0, K, lane);
        v16h bf1 = load_frag16(B1 + k0, K, lane);
        v16h bf2 = load_frag16(B2 + k0, K, lane);
        v16h bf3 = load_frag16(B3 + k0, K, lane);
        acc0 = __builtin_amdgcn_wmma_f32_16x16x32_f16(false, af, false, bf0, (short)0, acc0, false, false);
        acc1 = __builtin_amdgcn_wmma_f32_16x16x32_f16(false, af, false, bf1, (short)0, acc1, false, false);
        acc2 = __builtin_amdgcn_wmma_f32_16x16x32_f16(false, af, false, bf2, (short)0, acc2, false, false);
        acc3 = __builtin_amdgcn_wmma_f32_16x16x32_f16(false, af, false, bf3, (short)0, acc3, false, false);
    }

    int ldc = ntiles * 16;
    int rowb = m_tile * 16 + ((lane & 16) ? 8 : 0);
    int colb = n0 * 16 + (lane & 15);
    #pragma unroll
    for (int r = 0; r < 8; ++r) {
        size_t rr = (size_t)(rowb + r) * ldc;
        C[rr + colb +  0] = acc0[r];
        C[rr + colb + 16] = acc1[r];
        C[rr + colb + 32] = acc2[r];
        C[rr + colb + 48] = acc3[r];
    }
}

// ---------------------------------------------------------------------------
// 3) depthwise causal conv(4) + SiLU over L: zx(:, 256..767) -> xbc(b,l,512)
// ---------------------------------------------------------------------------
__global__ void k_conv_silu(const float* __restrict__ zx,
                            const float* __restrict__ cw,
                            const float* __restrict__ cb,
                            float* __restrict__ xbc) {
    int row = blockIdx.x;                 // b*LSEQ + l
    int b = row >> 10, l = row & 1023;
    int c = threadIdx.x;                  // 0..511
    float acc = cb[c];
    #pragma unroll
    for (int j = 0; j < DCONV; ++j) {
        int ll = l - (DCONV - 1) + j;
        if (ll >= 0)
            acc += zx[((size_t)(b << 10) + ll) * DPROJ_P + DINNER + c] * cw[c * DCONV + j];
    }
    xbc[(size_t)row * CONVDIM + c] = silu_f(acc);
}

// ---------------------------------------------------------------------------
// 4) selective scan: one block per (b,head); 256 threads
//    thread t owns p = t/4 (headdim row) and n in [(t&3)*32, +32) of the state
// ---------------------------------------------------------------------------
__global__ void k_scan(const float* __restrict__ zx,
                       const float* __restrict__ xbc,
                       const float* __restrict__ dt_bias,
                       const float* __restrict__ A_log,
                       const float* __restrict__ Dp,
                       float* __restrict__ ys) {
    __shared__ __align__(16) float sB[DSTATE];
    __shared__ __align__(16) float sC[DSTATE];
    int b = blockIdx.x >> 2;
    int h = blockIdx.x & 3;
    int t = threadIdx.x;
    int p  = t >> 2;
    int nb = (t & 3) * 32;

    float st[32];
    #pragma unroll
    for (int i = 0; i < 32; ++i) st[i] = 0.f;

    const float A   = -__expf(A_log[h]);
    const float dtb = dt_bias[h];
    const float dph = Dp[h];

    for (int l = 0; l < LSEQ; ++l) {
        size_t row = (size_t)(b << 10) + l;
        // prefetch next timestep's B/C/x rows (hide global latency behind FMAs)
        if (l + 1 < LSEQ && t < DSTATE) {
            __builtin_prefetch(&xbc[(row + 1) * CONVDIM + DINNER + 2 * t], 0, 0);
            if (t < HEADDIM)
                __builtin_prefetch(&xbc[(row + 1) * CONVDIM + h * HEADDIM + t], 0, 0);
        }
        if (t < DSTATE)       sB[t]           = xbc[row * CONVDIM + DINNER + t];
        else                  sC[t - DSTATE]  = xbc[row * CONVDIM + DINNER + DSTATE + (t - DSTATE)];
        float dt  = softplus_f(zx[row * DPROJ_P + (DPROJ - NHEADS) + h] + dtb);
        float dA  = __expf(dt * A);
        float x_p = xbc[row * CONVDIM + h * HEADDIM + p];
        __syncthreads();

        float coef = dt * x_p;
        float acc = 0.f;
        const float4* B4 = (const float4*)(sB + nb);
        const float4* C4 = (const float4*)(sC + nb);
        #pragma unroll
        for (int j = 0; j < 8; ++j) {
            float4 bv = B4[j], cv = C4[j];
            st[j*4+0] = st[j*4+0] * dA + coef * bv.x;  acc += st[j*4+0] * cv.x;
            st[j*4+1] = st[j*4+1] * dA + coef * bv.y;  acc += st[j*4+1] * cv.y;
            st[j*4+2] = st[j*4+2] * dA + coef * bv.z;  acc += st[j*4+2] * cv.z;
            st[j*4+3] = st[j*4+3] * dA + coef * bv.w;  acc += st[j*4+3] * cv.w;
        }
        acc += __shfl_xor(acc, 1, 32);
        acc += __shfl_xor(acc, 2, 32);
        if ((t & 3) == 0)
            ys[row * DINNER + h * HEADDIM + p] = acc + dph * x_p;
        __syncthreads();
    }
}

// ---------------------------------------------------------------------------
// 5) gate with silu(z), RMSNorm, -> yn_f16
// ---------------------------------------------------------------------------
__global__ void k_gate_rms(const float* __restrict__ ys,
                           const float* __restrict__ zx,
                           const float* __restrict__ rms_w,
                           _Float16* __restrict__ yn16) {
    __shared__ float sm[8];
    size_t row = blockIdx.x;
    int c = threadIdx.x;
    float y = ys[row * DINNER + c];
    float z = zx[row * DPROJ_P + c];
    y *= silu_f(z);
    float ss = block_reduce_sum256(y * y, sm) * (1.f / DINNER);
    float yn = y * rsqrtf(ss + EPSV) * rms_w[c];
    yn16[row * DINNER + c] = (_Float16)yn;
}

// ---------------------------------------------------------------------------
// 6) pooled[b,c] = mean_l outp[b,l,c]
// ---------------------------------------------------------------------------
__global__ void k_pool(const float* __restrict__ outp, float* __restrict__ pooled) {
    __shared__ float sm[8];
    int c = blockIdx.x;        // 0..255
    int b = blockIdx.y;        // 0..3
    float s = 0.f;
    for (int l = threadIdx.x; l < LSEQ; l += 256)
        s += outp[((size_t)(b << 10) + l) * DMODEL + c];
    s = block_reduce_sum256(s, sm);
    if (threadIdx.x == 0) pooled[b * DMODEL + c] = s * (1.f / LSEQ);
}

// ---------------------------------------------------------------------------
// 7) SE MLP: attn[b,c] = sigmoid(se2(relu(se1(pooled))))
// ---------------------------------------------------------------------------
__global__ void k_se(const float* __restrict__ pooled,
                     const float* __restrict__ w1, const float* __restrict__ b1,
                     const float* __restrict__ w2, const float* __restrict__ b2,
                     float* __restrict__ attn) {
    __shared__ float sp[DMODEL];
    __shared__ float sh[DMODEL / 4];
    int b = blockIdx.x, t = threadIdx.x;
    sp[t] = pooled[b * DMODEL + t];
    __syncthreads();
    if (t < DMODEL / 4) {
        float a = b1[t];
        for (int c = 0; c < DMODEL; ++c) a += sp[c] * w1[t * DMODEL + c];
        sh[t] = fmaxf(a, 0.f);
    }
    __syncthreads();
    float a = b2[t];
    for (int i = 0; i < DMODEL / 4; ++i) a += sh[i] * w2[t * (DMODEL / 4) + i];
    attn[b * DMODEL + t] = sigm_f(a);
}

// ---------------------------------------------------------------------------
// 8) final residual: out(b,c,h,w) = x + outp(b,l,c)*attn(b,c)
// ---------------------------------------------------------------------------
__global__ void k_final(const float* __restrict__ x,
                        const float* __restrict__ outp,
                        const float* __restrict__ attn,
                        float* __restrict__ out) {
    size_t i = (size_t)blockIdx.x * blockDim.x + threadIdx.x;
    if (i >= (size_t)BATCH * DMODEL * LSEQ) return;
    int l = i & 1023;
    int c = (i >> 10) & (DMODEL - 1);
    int b = i >> 18;
    out[i] = x[i] + outp[((size_t)(b << 10) + l) * DMODEL + c] * attn[b * DMODEL + c];
}

// ---------------------------------------------------------------------------
extern "C" void kernel_launch(void* const* d_in, const int* in_sizes, int n_in,
                              void* d_out, int out_size, void* d_ws, size_t ws_size,
                              hipStream_t stream) {
    const float* x       = (const float*)d_in[0];
    const float* norm_g  = (const float*)d_in[1];
    const float* norm_b  = (const float*)d_in[2];
    const float* in_w    = (const float*)d_in[3];
    const float* conv_w  = (const float*)d_in[4];
    const float* conv_b  = (const float*)d_in[5];
    const float* dt_bias = (const float*)d_in[6];
    const float* A_log   = (const float*)d_in[7];
    const float* Dp      = (const float*)d_in[8];
    const float* rms_w   = (const float*)d_in[9];
    const float* out_w   = (const float*)d_in[10];
    const float* se1_w   = (const float*)d_in[11];
    const float* se1_b   = (const float*)d_in[12];
    const float* se2_w   = (const float*)d_in[13];
    const float* se2_b   = (const float*)d_in[14];
    float* out = (float*)d_out;

    const size_t M = (size_t)BATCH * LSEQ;   // 4096
    char* ws = (char*)d_ws;
    size_t off = 0;
    auto alloc = [&](size_t bytes) { void* p = ws + off; off += (bytes + 255) & ~(size_t)255; return p; };

    _Float16* u16   = (_Float16*)alloc(M * DMODEL * 2);
    _Float16* w1_16 = (_Float16*)alloc((size_t)DPROJ_P * DMODEL * 2);
    _Float16* w2_16 = (_Float16*)alloc((size_t)DMODEL * DINNER * 2);
    float*    zx    = (float*)   alloc(M * DPROJ_P * 4);
    float*    xbc   = (float*)   alloc(M * CONVDIM * 4);
    float*    ys    = (float*)   alloc(M * DINNER * 4);
    _Float16* yn16  = (_Float16*)alloc(M * DINNER * 2);
    float*    outp  = (float*)   alloc(M * DMODEL * 4);
    float*    pooled= (float*)   alloc(BATCH * DMODEL * 4);
    float*    attn  = (float*)   alloc(BATCH * DMODEL * 4);
    (void)ws_size; (void)n_in; (void)in_sizes; (void)out_size;

    // weight conversion (padded for in_w: 772 -> 832 rows, pad rows are zero)
    {
        int nsrc = DPROJ * DMODEL, ndst = DPROJ_P * DMODEL;
        k_cvt_pad_f16<<<(ndst + 255) / 256, 256, 0, stream>>>(in_w, w1_16, nsrc, ndst);
        int n2 = DMODEL * DINNER;
        k_cvt_pad_f16<<<(n2 + 255) / 256, 256, 0, stream>>>(out_w, w2_16, n2, n2);
    }

    // 1) layernorm
    k_layernorm<<<(int)M, DMODEL, 0, stream>>>(x, norm_g, norm_b, u16);

    // 2) in-proj GEMM: zx(4096 x 832) = u16(4096x256) * w1_16(832x256)^T
    //    52 n-tiles -> 13 groups of 4 -> grid.y = ceil(13/8) = 2
    k_wmma_gemm4<<<dim3(M / 16, 2), 256, 0, stream>>>(u16, w1_16, zx, DMODEL, DPROJ_P / 16);

    // 3) conv + silu
    k_conv_silu<<<(int)M, CONVDIM, 0, stream>>>(zx, conv_w, conv_b, xbc);

    // 4) scan: 16 blocks (b,head)
    k_scan<<<BATCH * NHEADS, 256, 0, stream>>>(zx, xbc, dt_bias, A_log, Dp, ys);

    // 5) gate + rmsnorm -> yn16
    k_gate_rms<<<(int)M, DINNER, 0, stream>>>(ys, zx, rms_w, yn16);

    // 6) out-proj GEMM: outp(4096 x 256) = yn16 * w2_16^T ; 16 tiles -> 4 groups
    k_wmma_gemm4<<<dim3(M / 16, 1), 256, 0, stream>>>(yn16, w2_16, outp, DINNER, DMODEL / 16);

    // 7) SE path
    k_pool<<<dim3(DMODEL, BATCH), 256, 0, stream>>>(outp, pooled);
    k_se<<<BATCH, DMODEL, 0, stream>>>(pooled, se1_w, se1_b, se2_w, se2_b, attn);

    // 8) residual
    size_t total = (size_t)BATCH * DMODEL * LSEQ;
    k_final<<<(int)((total + 255) / 256), 256, 0, stream>>>(x, outp, attn, out);
}
